// CustomBSplineLayer_45243185496820
// MI455X (gfx1250) — matrix operand order repaired and checked by reading, not compiled, not run on hardware
//
#include <hip/hip_runtime.h>

typedef __attribute__((ext_vector_type(16))) _Float16 v16h;
typedef __attribute__((ext_vector_type(8)))  _Float16 v8h;
typedef __attribute__((ext_vector_type(8)))  float    v8f;
typedef __attribute__((ext_vector_type(4)))  int      v4i;

// Problem constants (from reference)
#define BATCH    4096
#define IN_SZ    1024
#define OUT_SZ   1024
#define K_TOTAL  8192

// GEMM tiling: block tile 128x256, 8 waves in 2x4 grid of 64x64 wave tiles
#define BM 128
#define BN 256
#define BK 64
#define APAD 72   // halves per LDS row: 64 + 8 pad = 144 bytes (16B-aligned chunks)
#define BPAD 72

// ---------------------------------------------------------------------------
// Async global->LDS copy (gfx1250 GLOBAL_LOAD_ASYNC_TO_LDS_B128, ASYNCcnt).
// clang prototype: void(v4i AS1*, v4i AS3*, imm int offset, imm int cpol)
// ---------------------------------------------------------------------------
#if __has_builtin(__builtin_amdgcn_global_load_async_to_lds_b128)
#define HAVE_ASYNC_LDS 1
typedef __attribute__((address_space(1))) v4i* glb_v4i_ptr;
typedef __attribute__((address_space(3))) v4i* lds_v4i_ptr;
__device__ __forceinline__ void async_copy_b128(const _Float16* g, _Float16* l)
{
    __builtin_amdgcn_global_load_async_to_lds_b128(
        (glb_v4i_ptr)g, (lds_v4i_ptr)l, 0, 0);
}
#else
#define HAVE_ASYNC_LDS 0
#endif

// ---------------------------------------------------------------------------
// Kernel 1: spline basis. One thread per (b, i) element of x.
// Writes 8 f16 basis values (16-byte v8h) to A workspace: A[b][i*8+g].
// Knots: CENTERS[j] = -1 + j*(2/7); ZOOM=1, PAN=0.
// ---------------------------------------------------------------------------
__global__ __launch_bounds__(256) void bspline_basis_kernel(
    const float* __restrict__ x, _Float16* __restrict__ A, int total)
{
    int idx = blockIdx.x * blockDim.x + threadIdx.x;
    if (idx >= total) return;

    const float h  = 2.0f / 7.0f;
    const float c0 = -1.0f;
    float xt = tanhf(x[idx]);

    float b[11];
#pragma unroll
    for (int j = 0; j < 11; ++j) {
        float cj  = c0 + (float)j * h;
        float cj1 = c0 + (float)(j + 1) * h;
        b[j] = (xt >= cj && xt < cj1) ? 1.0f : 0.0f;
    }
#pragma unroll
    for (int k = 1; k <= 3; ++k) {
        int n = 12 - k - 1;
        float inv = 1.0f / ((float)k * h);
#pragma unroll
        for (int j = 0; j < n; ++j) {
            float cj   = c0 + (float)j * h;
            float cjk1 = c0 + (float)(j + k + 1) * h;
            b[j] = (xt - cj) * inv * b[j] + (cjk1 - xt) * inv * b[j + 1];
        }
    }

    v8h out;
#pragma unroll
    for (int j = 0; j < 8; ++j) out[j] = (_Float16)b[j];
    *(v8h*)(A + (size_t)idx * 8) = out;
}

// ---------------------------------------------------------------------------
// Kernel 2: coef f32 -> f16. coef's (o,i,g) flat layout is already the
// K-contiguous B^T layout: Bt[o][k], k = i*8+g.
// ---------------------------------------------------------------------------
__global__ __launch_bounds__(256) void coef_to_f16_kernel(
    const float* __restrict__ coef, _Float16* __restrict__ Bt, int total8)
{
    int idx = blockIdx.x * blockDim.x + threadIdx.x;
    if (idx >= total8) return;
    const float* src = coef + (size_t)idx * 8;
    v8h out;
#pragma unroll
    for (int j = 0; j < 8; ++j) out[j] = (_Float16)src[j];
    *(v8h*)(Bt + (size_t)idx * 8) = out;
}

// ---------------------------------------------------------------------------
// Kernel 3: WMMA GEMM.  C[M,N] = A[M,K] * Bt[N,K]^T  (NT layout).
// 256 threads = 8 waves; block tile 128x256; wave grid 2x4 of 64x64 wave
// tiles (4x4 WMMA tiles each). K advanced in BK=64 steps, LDS double-buffered
// with async global->LDS copies overlapping the WMMA phase.
// LDS: sA 2*128*72*2B = 36 KB, sB 2*256*72*2B = 72 KB  -> 108 KB (< 320 KB).
// ---------------------------------------------------------------------------
__global__ __launch_bounds__(256) void bspline_gemm_wmma_kernel(
    const _Float16* __restrict__ A,   // [BATCH][K_TOTAL]
    const _Float16* __restrict__ Bt,  // [OUT_SZ][K_TOTAL]
    float* __restrict__ C)            // [BATCH][OUT_SZ]
{
    __shared__ __align__(16) _Float16 sA[2][BM * APAD];
    __shared__ __align__(16) _Float16 sB[2][BN * BPAD];

    const int tid  = threadIdx.x;
    const int wave = tid >> 5;
    const int lane = tid & 31;
    const int wm   = wave >> 2;        // 0..1 : 64-row group
    const int wn   = wave & 3;         // 0..3 : 64-col group
    const int lane16   = lane & 15;
    const int half_sel = lane >> 4;

    const int block_m = blockIdx.y * BM;
    const int block_n = blockIdx.x * BN;

    // Tile fill:
    //   A tile 128x64: thread t owns 64B (4 x b128): row t>>1, half-row (t&1)
    //   B tile 256x64: thread t owns the full 128B row t (8 x b128)
    const int lrow = tid >> 1;             // 0..127
    const int lcol = (tid & 1) * 32;       // 0 or 32 halves

    const _Float16* gA = A  + (size_t)(block_m + lrow) * K_TOTAL + lcol;
    const _Float16* gB = Bt + (size_t)(block_n + tid) * K_TOTAL;

    v8f acc[4][4] = {};

    // ---- compute one BK-wide LDS round from buffer `buf` -------------------
    auto compute = [&](int buf) {
#pragma unroll
        for (int ks = 0; ks < BK; ks += 32) {
            v16h bfrag[4];
#pragma unroll
            for (int nt = 0; nt < 4; ++nt) {
                // lane holds Bt[n][ks + h*16 + (0..15)]
                const _Float16* p =
                    &sB[buf][(wn * 64 + nt * 16 + lane16) * BPAD + ks + half_sel * 16];
                v8h lo = *(const v8h*)p;
                v8h hi = *(const v8h*)(p + 8);
                bfrag[nt] = __builtin_shufflevector(
                    lo, hi, 0, 1, 2, 3, 4, 5, 6, 7, 8, 9, 10, 11, 12, 13, 14, 15);
            }
#pragma unroll
            for (int mt = 0; mt < 4; ++mt) {
                // lane holds A[m][ks + h*16 + (0..15)]
                const _Float16* p =
                    &sA[buf][(wm * 64 + mt * 16 + lane16) * APAD + ks];
                v8h lo = *(const v8h*)(p + half_sel * 16);
                v8h hi = *(const v8h*)(p + half_sel * 16 + 8);
                v16h afrag = __builtin_shufflevector(
                    lo, hi, 0, 1, 2, 3, 4, 5, 6, 7, 8, 9, 10, 11, 12, 13, 14, 15);
#pragma unroll
                for (int nt = 0; nt < 4; ++nt) {
                    acc[mt][nt] = __builtin_amdgcn_wmma_f32_16x16x32_f16(
                        false, afrag, false, bfrag[nt],
                        (short)0, acc[mt][nt], false, false);
                }
            }
        }
    };

    const int NIT = K_TOTAL / BK;   // 128

#if HAVE_ASYNC_LDS
    // ---- double-buffered pipeline with async global->LDS copies -----------
    auto fill = [&](int buf, int k0) {
        _Float16* la = &sA[buf][lrow * APAD + lcol];
        _Float16* lb = &sB[buf][tid * BPAD];
        const _Float16* ga = gA + k0;
        const _Float16* gb = gB + k0;
        async_copy_b128(ga,      la);
        async_copy_b128(ga + 8,  la + 8);
        async_copy_b128(ga + 16, la + 16);
        async_copy_b128(ga + 24, la + 24);
        async_copy_b128(gb,      lb);
        async_copy_b128(gb + 8,  lb + 8);
        async_copy_b128(gb + 16, lb + 16);
        async_copy_b128(gb + 24, lb + 24);
        async_copy_b128(gb + 32, lb + 32);
        async_copy_b128(gb + 40, lb + 40);
        async_copy_b128(gb + 48, lb + 48);
        async_copy_b128(gb + 56, lb + 56);
    };

    fill(0, 0);                                     // ASYNCcnt = 12
    for (int it = 0; it < NIT; ++it) {
        const int cur = it & 1;
        if (it + 1 < NIT) {
            fill(cur ^ 1, (it + 1) * BK);           // ASYNCcnt -> 24
            // in-order completion: <=12 outstanding => current tile landed
            asm volatile("s_wait_asynccnt 12" ::: "memory");
        } else {
            asm volatile("s_wait_asynccnt 0" ::: "memory");
        }
        __syncthreads();        // all waves' copies for `cur` complete
        compute(cur);
        __syncthreads();        // everyone done reading `cur` before refill
    }
#else
    // ---- synchronous fallback -------------------------------------------
    for (int it = 0; it < NIT; ++it) {
        const int k0 = it * BK;
        v8h a0 = *(const v8h*)(gA + k0);
        v8h a1 = *(const v8h*)(gA + k0 + 8);
        v8h a2 = *(const v8h*)(gA + k0 + 16);
        v8h a3 = *(const v8h*)(gA + k0 + 24);
        v8h b0 = *(const v8h*)(gB + k0);
        v8h b1 = *(const v8h*)(gB + k0 + 8);
        v8h b2 = *(const v8h*)(gB + k0 + 16);
        v8h b3 = *(const v8h*)(gB + k0 + 24);
        v8h b4 = *(const v8h*)(gB + k0 + 32);
        v8h b5 = *(const v8h*)(gB + k0 + 40);
        v8h b6 = *(const v8h*)(gB + k0 + 48);
        v8h b7 = *(const v8h*)(gB + k0 + 56);
        __syncthreads();
        _Float16* la = &sA[0][lrow * APAD + lcol];
        _Float16* lb = &sB[0][tid * BPAD];
        *(v8h*)(la)      = a0; *(v8h*)(la + 8)  = a1;
        *(v8h*)(la + 16) = a2; *(v8h*)(la + 24) = a3;
        *(v8h*)(lb)      = b0; *(v8h*)(lb + 8)  = b1;
        *(v8h*)(lb + 16) = b2; *(v8h*)(lb + 24) = b3;
        *(v8h*)(lb + 32) = b4; *(v8h*)(lb + 40) = b5;
        *(v8h*)(lb + 48) = b6; *(v8h*)(lb + 56) = b7;
        __syncthreads();
        compute(0);
    }
#endif

    // ---- epilogue: C layout per lane: VGPR r -> row r + 8*h, col lane16 ----
#pragma unroll
    for (int mt = 0; mt < 4; ++mt) {
#pragma unroll
        for (int nt = 0; nt < 4; ++nt) {
            const int n     = block_n + wn * 64 + nt * 16 + lane16;
            const int mbase = block_m + wm * 64 + mt * 16 + half_sel * 8;
#pragma unroll
            for (int r = 0; r < 8; ++r) {
                C[(size_t)(mbase + r) * OUT_SZ + n] = acc[mt][nt][r];
            }
        }
    }
}

// ---------------------------------------------------------------------------
extern "C" void kernel_launch(void* const* d_in, const int* in_sizes, int n_in,
                              void* d_out, int out_size, void* d_ws, size_t ws_size,
                              hipStream_t stream)
{
    const float* x    = (const float*)d_in[0];   // 4096*1024
    const float* coef = (const float*)d_in[1];   // 1024*1024*8
    float* out = (float*)d_out;                  // 4096*1024

    _Float16* Aws = (_Float16*)d_ws;                                         // 64 MB
    _Float16* Bws = (_Float16*)((char*)d_ws +
                                (size_t)BATCH * K_TOTAL * sizeof(_Float16)); // +16 MB

    {
        int total = BATCH * IN_SZ;
        hipLaunchKernelGGL(bspline_basis_kernel,
                           dim3((total + 255) / 256), dim3(256), 0, stream,
                           x, Aws, total);
    }
    {
        int total8 = (OUT_SZ * K_TOTAL) / 8;
        hipLaunchKernelGGL(coef_to_f16_kernel,
                           dim3((total8 + 255) / 256), dim3(256), 0, stream,
                           coef, Bws, total8);
    }
    {
        dim3 grid(OUT_SZ / BN, BATCH / BM);   // (4, 32)
        hipLaunchKernelGGL(bspline_gemm_wmma_kernel,
                           grid, dim3(256), 0, stream,
                           Aws, Bws, out);
    }
}
